// SelfAttentionv2_13700945674463
// MI455X (gfx1250) — matrix operand order repaired and checked by reading, not compiled, not run on hardware
//
#include <hip/hip_runtime.h>

// ---------------------------------------------------------------------------
// Self-attention (N=4096, d_in=d_out=1024), fp32 in/out, f16 WMMA compute.
//   q = x@Wq * (1/sqrt(d)); k = x@Wk; v = x@Wv
//   out = softmax(q@k^T) @ v      (flash-style online softmax)
//
// Layout strategy (all WMMA operands are contiguous b128 loads):
//   x, Q, K : row-major  [N x D]   (A-fragments / K^T B-fragments: contiguous)
//   Wq/k/v  : TRANSPOSED [D_out x D_in] f16 (B-fragments: contiguous)
//   V       : TRANSPOSED [D x N]   f16      (B-fragments: contiguous)
// ---------------------------------------------------------------------------

#define D_DIM 1024
#define N_DIM 4096

typedef __attribute__((ext_vector_type(16))) _Float16 v16h;
typedef __attribute__((ext_vector_type(8)))  _Float16 v8h;
typedef __attribute__((ext_vector_type(8)))  float    v8f;
typedef __attribute__((ext_vector_type(4)))  float    v4f;

__device__ __forceinline__ v16h cat8(v8h lo, v8h hi) {
  return __builtin_shufflevector(lo, hi, 0, 1, 2, 3, 4, 5, 6, 7,
                                         8, 9, 10, 11, 12, 13, 14, 15);
}

// --- A-fragment: 16x32 f16 tile of row-major [*, ld] matrix at (row0, k0).
__device__ __forceinline__ v16h load_a_frag(const _Float16* __restrict__ src,
                                            int ld, int row0, int k0) {
  const int lane = threadIdx.x & 31;
  const int g = lane >> 4;
  const _Float16* p = src + (size_t)(row0 + (lane & 15)) * ld + k0 + g * 8;
  v8h lo = *(const v8h*)(p);
  v8h hi = *(const v8h*)(p + 16);
  return cat8(lo, hi);
}

// --- B-fragment: 32x16 where B[k][n] = src[(row0+n)*ld + k0+k]
// (source holds the B operand TRANSPOSED, row-major; K = 8g..8g+7, 16+8g..16+8g+7 per lane).
__device__ __forceinline__ v16h load_bT_frag(const _Float16* __restrict__ src,
                                             int ld, int row0, int k0) {
  const int lane = threadIdx.x & 31;
  const _Float16* p =
      src + (size_t)(row0 + (lane & 15)) * ld + k0 + (lane >> 4) * 8;
  v8h lo = *(const v8h*)(p);
  v8h hi = *(const v8h*)(p + 16);
  return cat8(lo, hi);
}

// --- store 16x16 f32 C/D tile as f16, row-major (VGPR r -> rows r / r+8)
__device__ __forceinline__ void store_c_f16(_Float16* __restrict__ dst, int ld,
                                            int row0, int n0, v8f c, float scale) {
  const int lane = threadIdx.x & 31;
  const int g = lane >> 4;
#pragma unroll
  for (int r = 0; r < 8; ++r)
    dst[(size_t)(row0 + g * 8 + r) * ld + n0 + (lane & 15)] =
        (_Float16)(c[r] * scale);
}

// --- store 16x16 f32 C/D tile as f16 into a TRANSPOSED destination
// dstT[n][row]: per lane the 8 row-dim elements are contiguous -> one b128.
__device__ __forceinline__ void store_c_f16_T(_Float16* __restrict__ dstT, int ldT,
                                              int row0, int n0, v8f c) {
  const int lane = threadIdx.x & 31;
  const int g = lane >> 4;
  v8h h;
#pragma unroll
  for (int r = 0; r < 8; ++r) h[r] = (_Float16)c[r];
  *(v8h*)(dstT + (size_t)(n0 + (lane & 15)) * ldT + row0 + g * 8) = h;
}

// ---------------------------------------------------------------------------
// 8-wide f32 -> f16 convert (b128 in/out)
__global__ void cvt_x_kernel(const float* __restrict__ src,
                             _Float16* __restrict__ dst, int n) {
  int i = (blockIdx.x * blockDim.x + threadIdx.x) * 8;
  if (i < n) {
    v4f a = *(const v4f*)(src + i);
    v4f b = *(const v4f*)(src + i + 4);
    v8h h;
#pragma unroll
    for (int e = 0; e < 4; ++e) { h[e] = (_Float16)a[e]; h[e + 4] = (_Float16)b[e]; }
    *(v8h*)(dst + i) = h;
  }
}

// f32 [D x D] row-major -> f16 transposed [D x D] (coalesced reads, one-time)
__global__ void cvt_wT_kernel(const float* __restrict__ src,
                              _Float16* __restrict__ dstT) {
  int i = blockIdx.x * blockDim.x + threadIdx.x;  // i < D*D
  int row = i >> 10;          // d_in index
  int col = i & (D_DIM - 1);  // d_out index
  dstT[(size_t)col * D_DIM + row] = (_Float16)src[i];
}

// ---------------------------------------------------------------------------
// Fused QKV projection: one wave per 32x16 output strip (2 A-frags x 3 B-frags
// = 6 WMMAs per k-step from 10 b128 loads). Scale 1/sqrt(D) folded into Q.
// V is stored transposed for the attention-side B-fragments.
__global__ __launch_bounds__(32) void qkv_proj_kernel(
    const _Float16* __restrict__ xh,
    const _Float16* __restrict__ wqt, const _Float16* __restrict__ wkt,
    const _Float16* __restrict__ wvt,
    _Float16* __restrict__ Q, _Float16* __restrict__ K, _Float16* __restrict__ Vt) {
  const int m0 = blockIdx.x * 32;
  const int n0 = blockIdx.y * 16;
  const v8f vzero = {0.f, 0.f, 0.f, 0.f, 0.f, 0.f, 0.f, 0.f};
  v8f aq0 = vzero, aq1 = vzero, ak0 = vzero, ak1 = vzero, av0 = vzero, av1 = vzero;
#pragma unroll 2
  for (int k0 = 0; k0 < D_DIM; k0 += 32) {
    v16h a0 = load_a_frag(xh, D_DIM, m0, k0);
    v16h a1 = load_a_frag(xh, D_DIM, m0 + 16, k0);
    v16h bq = load_bT_frag(wqt, D_DIM, n0, k0);
    v16h bk = load_bT_frag(wkt, D_DIM, n0, k0);
    v16h bv = load_bT_frag(wvt, D_DIM, n0, k0);
    aq0 = __builtin_amdgcn_wmma_f32_16x16x32_f16(false, a0, false, bq, (short)0, aq0, false, false);
    aq1 = __builtin_amdgcn_wmma_f32_16x16x32_f16(false, a1, false, bq, (short)0, aq1, false, false);
    ak0 = __builtin_amdgcn_wmma_f32_16x16x32_f16(false, a0, false, bk, (short)0, ak0, false, false);
    ak1 = __builtin_amdgcn_wmma_f32_16x16x32_f16(false, a1, false, bk, (short)0, ak1, false, false);
    av0 = __builtin_amdgcn_wmma_f32_16x16x32_f16(false, a0, false, bv, (short)0, av0, false, false);
    av1 = __builtin_amdgcn_wmma_f32_16x16x32_f16(false, a1, false, bv, (short)0, av1, false, false);
  }
  store_c_f16(Q, D_DIM, m0,      n0, aq0, 0.03125f);  // 1/sqrt(1024)
  store_c_f16(Q, D_DIM, m0 + 16, n0, aq1, 0.03125f);
  store_c_f16(K, D_DIM, m0,      n0, ak0, 1.0f);
  store_c_f16(K, D_DIM, m0 + 16, n0, ak1, 1.0f);
  store_c_f16_T(Vt, N_DIM, m0,      n0, av0);
  store_c_f16_T(Vt, N_DIM, m0 + 16, n0, av1);
}

// ---------------------------------------------------------------------------
// Flash attention: block = 256 threads (8 waves) handles 16 query rows.
// Wave w owns output columns [w*128, w*128+128) and the matching 128-wide
// d-slice for partial score computation. V-fragments are preloaded before
// the softmax barriers so their latency hides behind the LDS reduction.
__global__ __launch_bounds__(256) void attn_kernel(
    const _Float16* __restrict__ Q, const _Float16* __restrict__ Kh,
    const _Float16* __restrict__ Vt, float* __restrict__ O) {
  __shared__ float    Sred[16][32];   // cross-wave score reduction
  __shared__ _Float16 Pb[16][32];     // softmax probabilities (f16)
  __shared__ float    mrow[16], lrow[16], arow[16];

  const int tid  = threadIdx.x;
  const int wave = tid >> 5;
  const int lane = tid & 31;
  const int g    = lane >> 4;
  const int q0   = blockIdx.x * 16;
  const int col0 = wave * 128;

  if (tid < 16) { mrow[tid] = -3.0e38f; lrow[tid] = 0.0f; }

  // Q fragments for this wave's d-slice stay in registers for the whole kernel
  v16h qf[4];
#pragma unroll
  for (int i = 0; i < 4; ++i) qf[i] = load_a_frag(Q, D_DIM, q0, col0 + i * 32);

  const v8f vzero = {0.f, 0.f, 0.f, 0.f, 0.f, 0.f, 0.f, 0.f};
  v8f acc[8];
#pragma unroll
  for (int c = 0; c < 8; ++c) acc[c] = vzero;

  __syncthreads();

  for (int key0 = 0; key0 < N_DIM; key0 += 32) {
    // prefetch next key block (K rows for this wave's d-slice)
    if (key0 + 32 < N_DIM) {
      __builtin_prefetch(Kh + (size_t)(key0 + 32 + lane) * D_DIM + col0, 0, 1);
    }

    // zero the reduction buffer (512 floats, 2 per thread)
    ((float*)Sred)[tid]       = 0.0f;
    ((float*)Sred)[tid + 256] = 0.0f;
    __syncthreads();

    // partial S(16x32) over this wave's 128-wide d-slice: 8 WMMAs
    v8f s0 = vzero, s1 = vzero;
#pragma unroll
    for (int i = 0; i < 4; ++i) {
      const int k0 = col0 + i * 32;
      v16h b0 = load_bT_frag(Kh, D_DIM, key0,      k0);
      v16h b1 = load_bT_frag(Kh, D_DIM, key0 + 16, k0);
      s0 = __builtin_amdgcn_wmma_f32_16x16x32_f16(false, qf[i], false, b0, (short)0, s0, false, false);
      s1 = __builtin_amdgcn_wmma_f32_16x16x32_f16(false, qf[i], false, b1, (short)0, s1, false, false);
    }

    // Preload all V B-fragments for this key block NOW: independent of the
    // softmax, so their global latency hides behind reduction + barriers.
    v16h vb[8];
#pragma unroll
    for (int c = 0; c < 8; ++c)
      vb[c] = load_bT_frag(Vt, N_DIM, col0 + c * 16, key0);

    // LDS f32 atomic reduction across the 8 waves (ds_add_f32)
#pragma unroll
    for (int r = 0; r < 8; ++r) {
      atomicAdd(&Sred[g * 8 + r][lane & 15],        s0[r]);
      atomicAdd(&Sred[g * 8 + r][16 + (lane & 15)], s1[r]);
    }
    __syncthreads();

    // online softmax: wave 0, lane = (row, column-half); shfl for row reduce
    if (wave == 0) {
      const int row = lane & 15;
      const int c0 = g * 16;
      float mx = -3.0e38f;
#pragma unroll
      for (int j = 0; j < 16; ++j) mx = fmaxf(mx, Sred[row][c0 + j]);
      mx = fmaxf(mx, __shfl_xor(mx, 16, 32));
      mx = fmaxf(mx, mrow[row]);
      const float alpha = __expf(mrow[row] - mx);
      float l = 0.0f;
#pragma unroll
      for (int j = 0; j < 16; ++j) {
        const float p = __expf(Sred[row][c0 + j] - mx);
        Pb[row][c0 + j] = (_Float16)p;
        l += p;
      }
      l += __shfl_xor(l, 16, 32);
      if (g == 0) {
        lrow[row] = lrow[row] * alpha + l;
        mrow[row] = mx;
        arow[row] = alpha;
      }
    }
    __syncthreads();

    // rescale running output by alpha(row)
    float al[8];
#pragma unroll
    for (int r = 0; r < 8; ++r) al[r] = arow[g * 8 + r];
#pragma unroll
    for (int c = 0; c < 8; ++c)
#pragma unroll
      for (int r = 0; r < 8; ++r) acc[c][r] *= al[r];

    // P fragment (16x32 f16) from LDS — A layout, two b128 ds loads
    v16h pf;
    {
      const _Float16* p = &Pb[lane & 15][g * 8];
      v8h lo = *(const v8h*)(p);
      v8h hi = *(const v8h*)(p + 16);
      pf = cat8(lo, hi);
    }
    // O(16x128) += P(16x32) @ V(32x128) with the preloaded V fragments
#pragma unroll
    for (int c = 0; c < 8; ++c)
      acc[c] = __builtin_amdgcn_wmma_f32_16x16x32_f16(false, pf, false, vb[c], (short)0, acc[c], false, false);
    // Pb reads are protected by the zero-phase barrier of the next iteration
  }

  // normalize by row sums and store fp32
  float li[8];
#pragma unroll
  for (int r = 0; r < 8; ++r) li[r] = 1.0f / lrow[g * 8 + r];
#pragma unroll
  for (int c = 0; c < 8; ++c) {
    const int n = col0 + c * 16 + (lane & 15);
#pragma unroll
    for (int r = 0; r < 8; ++r)
      O[(size_t)(q0 + g * 8 + r) * D_DIM + n] = acc[c][r] * li[r];
  }
}

// ---------------------------------------------------------------------------
extern "C" void kernel_launch(void* const* d_in, const int* in_sizes, int n_in,
                              void* d_out, int out_size, void* d_ws, size_t ws_size,
                              hipStream_t stream) {
  (void)in_sizes; (void)n_in; (void)out_size; (void)ws_size;
  const float* x  = (const float*)d_in[0];
  const float* Wq = (const float*)d_in[1];
  const float* Wk = (const float*)d_in[2];
  const float* Wv = (const float*)d_in[3];
  float* out = (float*)d_out;

  // f16 workspace layout (total 19M halfs = 38 MB)
  _Float16* xh   = (_Float16*)d_ws;
  _Float16* wqt  = xh   + (size_t)N_DIM * D_DIM;
  _Float16* wkt  = wqt  + (size_t)D_DIM * D_DIM;
  _Float16* wvt  = wkt  + (size_t)D_DIM * D_DIM;
  _Float16* Qh   = wvt  + (size_t)D_DIM * D_DIM;
  _Float16* Kh   = Qh   + (size_t)N_DIM * D_DIM;
  _Float16* Vth  = Kh   + (size_t)N_DIM * D_DIM;  // V transposed [D x N]

  const int nx = N_DIM * D_DIM, nw = D_DIM * D_DIM;
  cvt_x_kernel <<<nx / 8 / 256, 256, 0, stream>>>(x, xh, nx);
  cvt_wT_kernel<<<nw / 256, 256, 0, stream>>>(Wq, wqt);
  cvt_wT_kernel<<<nw / 256, 256, 0, stream>>>(Wk, wkt);
  cvt_wT_kernel<<<nw / 256, 256, 0, stream>>>(Wv, wvt);

  qkv_proj_kernel<<<dim3(N_DIM / 32, D_DIM / 16), 32, 0, stream>>>(
      xh, wqt, wkt, wvt, Qh, Kh, Vth);

  attn_kernel<<<N_DIM / 16, 256, 0, stream>>>(Qh, Kh, Vth, out);
}
